// EA_Module_76879914598492
// MI455X (gfx1250) — matrix operand order repaired and checked by reading, not compiled, not run on hardware
//
#include <hip/hip_runtime.h>
#include <hip/hip_fp16.h>

typedef float    v2f  __attribute__((ext_vector_type(2)));
typedef float    v8f  __attribute__((ext_vector_type(8)));
typedef _Float16 v8h  __attribute__((ext_vector_type(8)));
typedef _Float16 v16h __attribute__((ext_vector_type(16)));

#define C_DIM  64
#define CK     8
#define NPOS   (96*96)      // 9216
#define BATCH  2
#define TILE_J 32           // K-dim per f16 WMMA in the PV matmul

static __device__ inline v8f v8f_zero() {
    v8f r;
#pragma unroll
    for (int i = 0; i < 8; ++i) r[i] = 0.0f;
    return r;
}

// ---------------------------------------------------------------------------
// Kernel 1: 1x1-conv projections.
//   qws: [B][N][8]  f32  (n-major -> contiguous b64 per-lane A/B fragment loads)
//   kws: [B][N][8]  f32
//   vws: [B][64][N] f16  (c-major -> contiguous 32B per-lane B fragment loads)
// ---------------------------------------------------------------------------
__global__ __launch_bounds__(256)
void qkv_proj_kernel(const float* __restrict__ x, const float* __restrict__ edge,
                     const float* __restrict__ Wq, const float* __restrict__ bq,
                     const float* __restrict__ Wk, const float* __restrict__ bk,
                     const float* __restrict__ Wv, const float* __restrict__ bv,
                     float* __restrict__ qws, float* __restrict__ kws,
                     _Float16* __restrict__ vws)
{
    __shared__ float sWq[CK * C_DIM], sWk[CK * C_DIM], sWv[C_DIM * C_DIM];
    __shared__ float sbq[CK], sbk[CK], sbv[C_DIM];

    for (int i = threadIdx.x; i < CK * C_DIM; i += blockDim.x) { sWq[i] = Wq[i]; sWk[i] = Wk[i]; }
    for (int i = threadIdx.x; i < C_DIM * C_DIM; i += blockDim.x) sWv[i] = Wv[i];
    if (threadIdx.x < CK)    { sbq[threadIdx.x] = bq[threadIdx.x]; sbk[threadIdx.x] = bk[threadIdx.x]; }
    if (threadIdx.x < C_DIM) sbv[threadIdx.x] = bv[threadIdx.x];
    __syncthreads();

    const int idx = blockIdx.x * blockDim.x + threadIdx.x;
    if (idx >= BATCH * NPOS) return;
    const int b = idx / NPOS, n = idx % NPOS;

    float qa[CK], ka[CK], va[C_DIM];
#pragma unroll
    for (int o = 0; o < CK; ++o) { qa[o] = sbq[o]; ka[o] = sbk[o]; }
#pragma unroll
    for (int o = 0; o < C_DIM; ++o) va[o] = sbv[o];

    const float* xb = x    + (size_t)b * C_DIM * NPOS + n;
    const float* eb = edge + (size_t)b * C_DIM * NPOS + n;
    for (int c = 0; c < C_DIM; ++c) {
        const float xv = xb[(size_t)c * NPOS];
        const float ev = eb[(size_t)c * NPOS];
#pragma unroll
        for (int o = 0; o < CK; ++o) {
            qa[o] += sWq[o * C_DIM + c] * ev;
            ka[o] += sWk[o * C_DIM + c] * xv;
        }
#pragma unroll
        for (int o = 0; o < C_DIM; ++o) va[o] += sWv[o * C_DIM + c] * xv;
    }

    float* qp = qws + ((size_t)b * NPOS + n) * CK;
    float* kp = kws + ((size_t)b * NPOS + n) * CK;
#pragma unroll
    for (int o = 0; o < CK; ++o) { qp[o] = qa[o]; kp[o] = ka[o]; }
    _Float16* vp = vws + (size_t)b * C_DIM * NPOS + n;
#pragma unroll
    for (int o = 0; o < C_DIM; ++o) vp[(size_t)o * NPOS] = (_Float16)va[o];
}

// ---------------------------------------------------------------------------
// Kernel 2: fused flash-attention. Block = 128 threads = 4 waves.
// Wave w handles 16 query rows starting at blockIdx.x*64 + w*16.
// S = Q^T K via V_WMMA_F32_16X16X4_F32, online softmax (shfl_xor across the
// two lane halves of wave32), then O += P*V via V_WMMA_F32_16X16X32_F16.
// ---------------------------------------------------------------------------
__global__ __launch_bounds__(128)
void flash_attn_kernel(const float* __restrict__ qws, const float* __restrict__ kws,
                       const _Float16* __restrict__ vws, const float* __restrict__ x,
                       const float* __restrict__ gamma, float* __restrict__ out)
{
    __shared__ float    Sbuf[4][16 * TILE_J];   // per-wave S tile (row-major f32)
    __shared__ _Float16 Pfrag[4][32 * 16];      // per-wave P tile, A-fragment order

    const int lane = threadIdx.x & 31;
    const int wave = threadIdx.x >> 5;
    const int lh   = lane >> 4;                 // lane half (0/1)
    const int lm   = lane & 15;
    const int b    = blockIdx.y;
    const int i0   = blockIdx.x * 64 + wave * 16;

    const float*    qb = qws + (size_t)b * NPOS * CK;
    const float*    kb = kws + (size_t)b * NPOS * CK;
    const _Float16* vb = vws + (size_t)b * C_DIM * NPOS;

    // Q A-fragment (16x4 f32 layout): a[r] = Q[i0+lm][d = r + 2*lh (+4 hi)]
    const float* qrow = qb + (size_t)(i0 + lm) * CK;
    const v2f aq_lo = *(const v2f*)(qrow + 2 * lh);
    const v2f aq_hi = *(const v2f*)(qrow + 4 + 2 * lh);

    v8f o0 = v8f_zero(), o1 = v8f_zero(), o2 = v8f_zero(), o3 = v8f_zero();
    float m_run = -INFINITY, l_run = 0.0f;

    float*    Sw = Sbuf[wave];
    _Float16* Pw = Pfrag[wave];

    for (int j0 = 0; j0 < NPOS; j0 += TILE_J) {
        // ---- K B-fragments (4x16 f32 layout): b[r] = K[j][d = r + 2*lh (+4 hi)]
        const float* k0 = kb + (size_t)(j0 + lm) * CK;
        const float* k1 = kb + (size_t)(j0 + 16 + lm) * CK;
        const v2f bk0_lo = *(const v2f*)(k0 + 2 * lh);
        const v2f bk0_hi = *(const v2f*)(k0 + 4 + 2 * lh);
        const v2f bk1_lo = *(const v2f*)(k1 + 2 * lh);
        const v2f bk1_hi = *(const v2f*)(k1 + 4 + 2 * lh);

        // ---- V B-fragments (32x16 f16): vb_t[h] = V[c = t*16+lm][j0 + h + 16*lh]
        // c-major layout -> one contiguous 32B load per fragment; issued early,
        // consumed only after the softmax (latency overlap).
        const v16h vb0 = *(const v16h*)(vb + (size_t)( 0 + lm) * NPOS + j0 + 16 * lh);
        const v16h vb1 = *(const v16h*)(vb + (size_t)(16 + lm) * NPOS + j0 + 16 * lh);
        const v16h vb2 = *(const v16h*)(vb + (size_t)(32 + lm) * NPOS + j0 + 16 * lh);
        const v16h vb3 = *(const v16h*)(vb + (size_t)(48 + lm) * NPOS + j0 + 16 * lh);

        // ---- prefetch next tile (global_prefetch_b8)
        if (j0 + TILE_J < NPOS) {
            __builtin_prefetch(kb + (size_t)(j0 + TILE_J + lm) * CK, 0, 0);
            __builtin_prefetch(vb + (size_t)lane * NPOS + j0 + TILE_J, 0, 0);
            __builtin_prefetch(vb + (size_t)(32 + lane) * NPOS + j0 + TILE_J, 0, 0);
        }

        // ---- S tile (16 x 32) = Q^T K : 4x V_WMMA_F32_16X16X4_F32 ----
        v8f s0 = v8f_zero(), s1 = v8f_zero();
        s0 = __builtin_amdgcn_wmma_f32_16x16x4_f32(false, aq_lo, false, bk0_lo, (short)0, s0, false, false);
        s0 = __builtin_amdgcn_wmma_f32_16x16x4_f32(false, aq_hi, false, bk0_hi, (short)0, s0, false, false);
        s1 = __builtin_amdgcn_wmma_f32_16x16x4_f32(false, aq_lo, false, bk1_lo, (short)0, s1, false, false);
        s1 = __builtin_amdgcn_wmma_f32_16x16x4_f32(false, aq_hi, false, bk1_hi, (short)0, s1, false, false);

        // ---- spill S (C/D layout) to LDS row-major ----
#pragma unroll
        for (int r = 0; r < 8; ++r) {
            const int i = r + 8 * lh;
            Sw[i * TILE_J + lm]      = s0[r];
            Sw[i * TILE_J + 16 + lm] = s1[r];
        }
        __syncthreads();

        // ---- online softmax: lane handles row lm, columns lh*16 .. lh*16+15 ----
        float pv[16];
        float rmax = -INFINITY;
#pragma unroll
        for (int c = 0; c < 16; ++c) {
            pv[c] = Sw[lm * TILE_J + lh * 16 + c];
            rmax  = fmaxf(rmax, pv[c]);
        }
        rmax = fmaxf(rmax, __shfl_xor(rmax, 16, 32));
        const float m_new = fmaxf(m_run, rmax);
        const float alpha = __expf(m_run - m_new);
        float psum = 0.0f;
#pragma unroll
        for (int c = 0; c < 16; ++c) {
            const float e = __expf(pv[c] - m_new);
            pv[c] = e;
            psum += e;
        }
        psum += __shfl_xor(psum, 16, 32);
        l_run = l_run * alpha + psum;
        m_run = m_new;

        // rescale accumulators: lane needs alpha of rows r + 8*lh (held by lane r+8*lh)
#pragma unroll
        for (int r = 0; r < 8; ++r) {
            const float ar = __shfl(alpha, r + 8 * lh, 32);
            o0[r] *= ar; o1[r] *= ar; o2[r] *= ar; o3[r] *= ar;
        }

        // ---- write P directly in A-fragment order (2x ds_store_b128 per lane):
        // element (row i, col c_abs) belongs to reader-lane (i + 16*(c_abs>>4))
        // at half-index h = c_abs & 15.
        v8h plo, phi;
#pragma unroll
        for (int c = 0; c < 8; ++c) {
            plo[c] = (_Float16)pv[c];
            phi[c] = (_Float16)pv[c + 8];
        }
        *(v8h*)&Pw[(lm + 16 * lh) * 16    ] = plo;   // c_abs = lh*16 + 0..7
        *(v8h*)&Pw[(lm + 16 * lh) * 16 + 8] = phi;   // c_abs = lh*16 + 8..15
        __syncthreads();

        // ---- gather P A-fragment: one contiguous 32B LDS read per lane ----
        const v16h pa = *(const v16h*)&Pw[lane * 16];

        // ---- O += P * V : 4x V_WMMA_F32_16X16X32_F16 ----
        o0 = __builtin_amdgcn_wmma_f32_16x16x32_f16(false, pa, false, vb0, (short)0, o0, false, false);
        o1 = __builtin_amdgcn_wmma_f32_16x16x32_f16(false, pa, false, vb1, (short)0, o1, false, false);
        o2 = __builtin_amdgcn_wmma_f32_16x16x32_f16(false, pa, false, vb2, (short)0, o2, false, false);
        o3 = __builtin_amdgcn_wmma_f32_16x16x32_f16(false, pa, false, vb3, (short)0, o3, false, false);
    }

    // ---- epilogue: normalize, gamma*O + x ----
    const float linv = 1.0f / l_run;          // denom for row lm (same in both halves)
    const float g    = gamma[0];
    float*       outb = out + (size_t)b * C_DIM * NPOS;
    const float* xb2  = x   + (size_t)b * C_DIM * NPOS;
#pragma unroll
    for (int r = 0; r < 8; ++r) {
        const float lr = __shfl(linv, r + 8 * lh, 32);
        const int   i  = i0 + r + 8 * lh;
        const size_t a0 = (size_t)( 0 + lm) * NPOS + i;
        const size_t a1 = (size_t)(16 + lm) * NPOS + i;
        const size_t a2 = (size_t)(32 + lm) * NPOS + i;
        const size_t a3 = (size_t)(48 + lm) * NPOS + i;
        outb[a0] = g * o0[r] * lr + xb2[a0];
        outb[a1] = g * o1[r] * lr + xb2[a1];
        outb[a2] = g * o2[r] * lr + xb2[a2];
        outb[a3] = g * o3[r] * lr + xb2[a3];
    }
}

extern "C" void kernel_launch(void* const* d_in, const int* in_sizes, int n_in,
                              void* d_out, int out_size, void* d_ws, size_t ws_size,
                              hipStream_t stream)
{
    (void)in_sizes; (void)n_in; (void)out_size; (void)ws_size;
    const float* x     = (const float*)d_in[0];
    const float* edge  = (const float*)d_in[1];
    const float* Wq    = (const float*)d_in[2];
    const float* bq    = (const float*)d_in[3];
    const float* Wk    = (const float*)d_in[4];
    const float* bk    = (const float*)d_in[5];
    const float* Wv    = (const float*)d_in[6];
    const float* bv    = (const float*)d_in[7];
    const float* gamma = (const float*)d_in[8];
    float* out = (float*)d_out;

    char*  ws  = (char*)d_ws;
    float* qws = (float*)ws;                                      // B*N*8 f32
    float* kws = qws + (size_t)BATCH * NPOS * CK;                 // B*N*8 f32
    _Float16* vws = (_Float16*)(kws + (size_t)BATCH * NPOS * CK); // B*64*N f16

    const int total = BATCH * NPOS;
    qkv_proj_kernel<<<(total + 255) / 256, 256, 0, stream>>>(
        x, edge, Wq, bq, Wk, bk, Wv, bv, qws, kws, vws);

    dim3 grid(NPOS / 64, BATCH);
    flash_attn_kernel<<<grid, 128, 0, stream>>>(qws, kws, vws, x, gamma, out);
}
